// TopoNet_57964878627368
// MI455X (gfx1250) — matrix operand order repaired and checked by reading, not compiled, not run on hardware
//
#include <hip/hip_runtime.h>

typedef __attribute__((ext_vector_type(16))) _Float16 v16h;
typedef __attribute__((ext_vector_type(8)))  _Float16 v8h;
typedef __attribute__((ext_vector_type(8)))  float    v8f;

// ---------------------------------------------------------------------------
// Weight repack for conv: OIHW f32 -> [CoutP][9][CinP] f16, zero-padded.
// CoutP = Mtiles*16, CinP = round_up(Cin,32).  Lets the conv A-fragment be two
// aligned 16B vector loads with no guards (pad contributes exact zeros).
// ---------------------------------------------------------------------------
__global__ void repack_w_kernel(const float* __restrict__ w, _Float16* __restrict__ w2,
                                int Cout, int Cin, int CinP, long total)
{
    for (long i = blockIdx.x * (long)blockDim.x + threadIdx.x; i < total;
         i += (long)gridDim.x * blockDim.x) {
        int ci = (int)(i % CinP);
        long r = i / CinP;
        int kk = (int)(r % 9);
        int co = (int)(r / 9);
        float v = (co < Cout && ci < Cin) ? w[((long)co * Cin + ci) * 9 + kk] : 0.f;
        w2[i] = (_Float16)v;
    }
}

// ---------------------------------------------------------------------------
// 3x3 implicit-GEMM conv via WMMA f32_16x16x32_f16, pad=1.
// GEMM view: M=Cout, N=B*OH*OW.  One 16x16 output tile per wave.
// (kh,kw) outer (ih/iw + base pointers hoisted), Cin chunks of 32 inner.
// Branchless hot loop: A = two 16B f16 vector loads from the zero-padded
// repack; B = 16 unconditional f32 loads at clamped addresses * sel (spatial
// zero-padding as data predication).  No div/mod, no exec-mask load branches.
// Fragment layouts used here (A and B share one contiguous k order):
// A-frag (16x32 f16): lane m = M%16, kbase=(lane<16?0:16), K(e)=kbase+e
// B-frag (32x16 f16): lane n = N%16, kbase=(lane<16?0:16), K(e)=kbase+e
// D (16x16 f32, 8 vgpr): M = e + (lane<16?0:8), N = lane%16
// ---------------------------------------------------------------------------
__global__ void conv3x3_wmma_kernel(const float* __restrict__ in,
                                    const _Float16* __restrict__ w2,  // CoutP x 9 x CinP
                                    const float* __restrict__ bias,
                                    float* __restrict__ out,
                                    int B, int Cin, int CinP, int H, int W,
                                    int Cout, int OH, int OW,
                                    int stride, int act,
                                    int Mtiles, int Ntiles)
{
    int wave = (int)((blockIdx.x * blockDim.x + threadIdx.x) >> 5);
    int lane = threadIdx.x & 31;
    if (wave >= Mtiles * Ntiles) return;           // wave-uniform

    const int mt = wave % Mtiles;
    const int nt = wave / Mtiles;
    const int m_base = mt * 16;
    const int n_base = nt * 16;
    const int Ntot = B * OH * OW;
    const long HWl = (long)H * W;

    const int lane16 = lane & 15;
    const int hi = lane >> 4;
    const int kbaseA = hi * 16;
    const int kbaseB = hi * 16;

    const int rowA = m_base + lane16;              // always < CoutP
    const _Float16* wrow = w2 + (long)rowA * 9 * CinP + kbaseA;

    const int nB = n_base + lane16;
    const bool nOk = nB < Ntot;
    int b = 0, oh = 0, ow = 0;
    if (nOk) {
        int sp = OH * OW;
        b = nB / sp; int r = nB % sp; oh = r / OW; ow = r % OW;
    }
    const float* inb = in + (long)b * Cin * HWl;   // batch base (b=0 for idle lanes)
    const int CinFull = Cin & ~31;

    v8f acc = {};
#pragma unroll
    for (int kh = 0; kh < 3; ++kh) {
#pragma unroll
        for (int kw = 0; kw < 3; ++kw) {
            const int ih = oh * stride - 1 + kh;
            const int iw = ow * stride - 1 + kw;
            const bool inOk = nOk && ih >= 0 && ih < H && iw >= 0 && iw < W;
            const float sel = inOk ? 1.f : 0.f;
            const int ihc = min(max(ih, 0), H - 1);
            const int iwc = min(max(iw, 0), W - 1);
            const float* bptr = inb + (long)ihc * W + iwc;       // + ci*HW
            const _Float16* wp = wrow + (kh * 3 + kw) * CinP;    // + ci (f16)

            int c0 = 0;
            for (; c0 < CinFull; c0 += 32) {
                if (c0 + 32 < CinFull)   // hide next chunk (global_prefetch_b8)
                    __builtin_prefetch(bptr + (long)(c0 + 32 + kbaseB) * HWl, 0, 1);

                v8h wlo = *(const v8h*)(wp + c0);        // ci = c0+kbA+0..7
                v8h whi = *(const v8h*)(wp + c0 + 8);    // ci = c0+kbA+8..15
                v16h afrag = __builtin_shufflevector(wlo, whi,
                    0, 1, 2, 3, 4, 5, 6, 7, 8, 9, 10, 11, 12, 13, 14, 15);

                float bv[16];
#pragma unroll
                for (int e = 0; e < 16; ++e)
                    bv[e] = bptr[(long)(c0 + kbaseB + e) * HWl];
                v16h bfrag;
#pragma unroll
                for (int e = 0; e < 16; ++e)
                    bfrag[e] = (_Float16)(bv[e] * sel);

                acc = __builtin_amdgcn_wmma_f32_16x16x32_f16(
                    false, afrag, false, bfrag, (short)0, acc, false, false);
            }
            if (c0 < Cin) {   // K tail: A pad is exact zero, so B only needs
                              // an in-bounds (clamped) address -- 0*finite==0
                v8h wlo = *(const v8h*)(wp + c0);
                v8h whi = *(const v8h*)(wp + c0 + 8);
                v16h afrag = __builtin_shufflevector(wlo, whi,
                    0, 1, 2, 3, 4, 5, 6, 7, 8, 9, 10, 11, 12, 13, 14, 15);

                float bv[16];
#pragma unroll
                for (int e = 0; e < 16; ++e) {
                    int ci = c0 + kbaseB + e;
                    bv[e] = bptr[(long)min(ci, Cin - 1) * HWl];
                }
                v16h bfrag;
#pragma unroll
                for (int e = 0; e < 16; ++e)
                    bfrag[e] = (_Float16)(bv[e] * sel);

                acc = __builtin_amdgcn_wmma_f32_16x16x32_f16(
                    false, afrag, false, bfrag, (short)0, acc, false, false);
            }
        }
    }
    if (!nOk) return;
#pragma unroll
    for (int e = 0; e < 8; ++e) {
        int m = m_base + e + hi * 8;
        if (m < Cout) {
            float v = acc[e] + bias[m];
            if (act == 1) v = fmaxf(v, 0.f);
            out[(((long)b * Cout + m) * OH + oh) * OW + ow] = v;
        }
    }
}

// ---------------------------------------------------------------------------
// Generic GEMM via WMMA:  out[n,m] = act( sum_k X[n,k] * Wm[m,k] + bias[m] )
// X: NxK row-major, Wm: MxK row-major, out: NxM row-major.
// Full-K chunks are branchless (clamped addresses; invalid rows/cols are
// never stored); only the K-tail uses select-by-multiply on A.
// ---------------------------------------------------------------------------
__global__ void gemm_wmma_kernel(const float* __restrict__ X,
                                 const float* __restrict__ Wm,
                                 const float* __restrict__ bias,
                                 float* __restrict__ out,
                                 int N, int K, int M, int act,
                                 int Mtiles, int Ntiles)
{
    int wave = (int)((blockIdx.x * blockDim.x + threadIdx.x) >> 5);
    int lane = threadIdx.x & 31;
    if (wave >= Mtiles * Ntiles) return;

    const int mt = wave % Mtiles;
    const int nt = wave / Mtiles;
    const int m_base = mt * 16;
    const int n_base = nt * 16;

    const int lane16 = lane & 15;
    const int hi = lane >> 4;
    const int kbaseA = hi * 8;
    const int kbaseB = hi * 16;

    const int rowA = m_base + lane16;
    const int nB = n_base + lane16;
    const bool nOk = nB < N;
    const float* wr = Wm + (long)min(rowA, M - 1) * K;   // clamped: never stored
    const float* xr = X + (long)min(nB, N - 1) * K;

    const int KFull = K & ~31;
    v8f acc = {};
    int k0 = 0;
    for (; k0 < KFull; k0 += 32) {
        v16h afrag;
#pragma unroll
        for (int e = 0; e < 16; ++e) {
            int k = k0 + kbaseB + e;
            afrag[e] = (_Float16)wr[k];
        }
        v16h bfrag;
#pragma unroll
        for (int e = 0; e < 16; ++e)
            bfrag[e] = (_Float16)xr[k0 + kbaseB + e];
        acc = __builtin_amdgcn_wmma_f32_16x16x32_f16(
            false, afrag, false, bfrag, (short)0, acc, false, false);
    }
    if (k0 < K) {   // tail: zero A for k>=K via select; clamp B address
        v16h afrag;
#pragma unroll
        for (int e = 0; e < 16; ++e) {
            int k = k0 + kbaseB + e;
            float v = wr[min(k, K - 1)] * ((k < K) ? 1.f : 0.f);
            afrag[e] = (_Float16)v;
        }
        v16h bfrag;
#pragma unroll
        for (int e = 0; e < 16; ++e) {
            int k = k0 + kbaseB + e;
            bfrag[e] = (_Float16)xr[min(k, K - 1)];
        }
        acc = __builtin_amdgcn_wmma_f32_16x16x32_f16(
            false, afrag, false, bfrag, (short)0, acc, false, false);
    }
    if (!nOk) return;
#pragma unroll
    for (int e = 0; e < 8; ++e) {
        int m = m_base + e + hi * 8;
        if (m < M) {
            float v = acc[e] + bias[m];
            if (act == 1) v = fmaxf(v, 0.f);
            out[(long)nB * M + m] = v;
        }
    }
}

// ---------------------------------------------------------------------------
// BatchNorm (training-mode batch stats): batch-outer / pixel-inner, no div/mod
// ---------------------------------------------------------------------------
__global__ void bn_stats_kernel(const float* __restrict__ x, float* __restrict__ stats,
                                int B, int C, int HW)
{
    int c = blockIdx.x;
    int tid = threadIdx.x;
    float s = 0.f, s2 = 0.f;
    for (int n = 0; n < B; ++n) {
        const float* p = x + ((long)n * C + c) * HW;
        for (int i = tid; i < HW; i += blockDim.x) {
            float v = p[i];
            s += v; s2 += v * v;
        }
    }
    __shared__ float sh[256], sh2[256];
    sh[tid] = s; sh2[tid] = s2; __syncthreads();
    for (int off = 128; off > 0; off >>= 1) {
        if (tid < off) { sh[tid] += sh[tid + off]; sh2[tid] += sh2[tid + off]; }
        __syncthreads();
    }
    if (tid == 0) {
        float total = (float)B * (float)HW;
        float mu = sh[0] / total;
        stats[c] = mu;
        stats[C + c] = sh2[0] / total - mu * mu;
    }
}

__global__ void bn_apply_kernel(float* __restrict__ x, const float* __restrict__ stats,
                                const float* __restrict__ gam, const float* __restrict__ bet,
                                int C, int HW, long total, int gelu)
{
    for (long i = blockIdx.x * (long)blockDim.x + threadIdx.x; i < total;
         i += (long)gridDim.x * blockDim.x) {
        int c = (int)((i / HW) % C);
        float mu = stats[c], var = stats[C + c];
        float v = (x[i] - mu) * rsqrtf(var + 1e-5f) * gam[c] + bet[c];
        if (gelu) v = 0.5f * v * (1.f + erff(v * 0.70710678118654752f));
        x[i] = v;
    }
}

// ---------------------------------------------------------------------------
// 1x1 conv to a single output channel (fw2 / sw2).  act: 0=none, 1=sigmoid
// ---------------------------------------------------------------------------
__global__ void conv1x1_kernel(const float* __restrict__ in, const float* __restrict__ w,
                               const float* __restrict__ bias, float* __restrict__ out,
                               int Cin, int HW, long BHW, int act)
{
    for (long i = blockIdx.x * (long)blockDim.x + threadIdx.x; i < BHW;
         i += (long)gridDim.x * blockDim.x) {
        long b = i / HW, p = i % HW;
        const float* ip = in + (long)b * Cin * HW + p;
        float s = bias[0];
        for (int c = 0; c < Cin; ++c)
            s += ip[(long)c * HW] * w[c];
        if (act == 1) s = 1.f / (1.f + expf(-s));
        out[i] = s;
    }
}

__global__ void minmax_kernel(const float* __restrict__ f, float* __restrict__ fmn,
                              float* __restrict__ fmx, int HW)
{
    int b = blockIdx.x, tid = threadIdx.x;
    const float* p = f + (long)b * HW;
    float mn = 3.4e38f, mx = -3.4e38f;
    for (int i = tid; i < HW; i += blockDim.x) {
        float v = p[i]; mn = fminf(mn, v); mx = fmaxf(mx, v);
    }
    __shared__ float smn[256], smx[256];
    smn[tid] = mn; smx[tid] = mx; __syncthreads();
    for (int off = 128; off > 0; off >>= 1) {
        if (tid < off) { smn[tid] = fminf(smn[tid], smn[tid + off]);
                         smx[tid] = fmaxf(smx[tid], smx[tid + off]); }
        __syncthreads();
    }
    if (tid == 0) { fmn[b] = smn[0]; fmx[b] = smx[0]; }
}

__global__ void levels_kernel(const float* __restrict__ off, float* __restrict__ lev, int n)
{
    if (blockIdx.x == 0 && threadIdx.x == 0) {
        float t[16];
        for (int i = 0; i < n; ++i) t[i] = 1.f / (1.f + expf(-off[i]));
        for (int i = 1; i < n; ++i) {
            float v = t[i]; int j = i - 1;
            while (j >= 0 && t[j] > v) { t[j + 1] = t[j]; --j; }
            t[j + 1] = v;
        }
        for (int i = 0; i < n; ++i) lev[i] = t[i];
    }
}

__global__ void fill_kernel(float* __restrict__ p, float v, long n)
{
    for (long i = blockIdx.x * (long)blockDim.x + threadIdx.x; i < n;
         i += (long)gridDim.x * blockDim.x) p[i] = v;
}

__global__ void mask_birth_kernel(const float* __restrict__ filt, float* __restrict__ maskb,
                                  float* __restrict__ birth, const float* __restrict__ lev,
                                  const float* __restrict__ fmn, const float* __restrict__ fmx,
                                  int lvl, int HW, long BHW)
{
    float L = lev[lvl];
    for (long i = blockIdx.x * (long)blockDim.x + threadIdx.x; i < BHW;
         i += (long)gridDim.x * blockDim.x) {
        long b = i / HW;
        float fr = fmaxf(fmx[b] - fmn[b], 1e-6f);
        float thr = fmn[b] + L * fr;
        float m = 1.f / (1.f + expf(-10.f * (thr - filt[i])));
        birth[i] += fmaxf(m - maskb[i], 0.f) * L;
        maskb[i] = m;
    }
}

// ---------------------------------------------------------------------------
// Fused diffusion for one filtration level: field = feat*mask, then 5 steps of
//   field = mask * (0.5*field + 0.5*neigh4(field)/max(neigh4(mask),0.25))
// all inside an LDS tile (32x32 output + halo 5), with the Welford mean/m2
// update fused at the end.  mask/cnt tiles loaded once, reused over channels.
// ---------------------------------------------------------------------------
__global__ void diffusion_kernel(const float* __restrict__ maskb,
                                 const float* __restrict__ feat,
                                 float* __restrict__ mean, float* __restrict__ m2,
                                 int C, int H, int W, float count, int cPerBlk)
{
    constexpr int TS = 32, HALO = 5, TW = TS + 2 * HALO;   // 42
    __shared__ float sm[TW * TW];
    __shared__ float sc[TW * TW];
    __shared__ float fA[TW * TW];
    __shared__ float fB[TW * TW];

    const int tilesX = (W + TS - 1) / TS;
    const int tx = (blockIdx.x % tilesX) * TS;
    const int ty = (blockIdx.x / tilesX) * TS;
    const int b = blockIdx.y;
    const int c0 = blockIdx.z * cPerBlk;
    const int tid = threadIdx.x, nth = blockDim.x;
    const int HW = H * W;

    const float* mb = maskb + (long)b * HW;
    for (int i = tid; i < TW * TW; i += nth) {
        int ly = i / TW, lx = i % TW;
        int gy = ty - HALO + ly, gx = tx - HALO + lx;
        sm[i] = (gy >= 0 && gy < H && gx >= 0 && gx < W) ? mb[gy * W + gx] : 0.f;
    }
    __syncthreads();
    for (int i = tid; i < TW * TW; i += nth) {
        int ly = i / TW, lx = i % TW;
        float cnt = 0.f;
        if (ly > 0 && ly < TW - 1 && lx > 0 && lx < TW - 1)
            cnt = 0.25f * (sm[i - TW] + sm[i + TW] + sm[i - 1] + sm[i + 1]);
        sc[i] = fmaxf(cnt, 0.25f);
    }
    __syncthreads();

    for (int c = c0; c < c0 + cPerBlk && c < C; ++c) {
        const float* fp = feat + ((long)(b * C + c)) * HW;
        for (int i = tid; i < TW * TW; i += nth) {
            int ly = i / TW, lx = i % TW;
            int gy = ty - HALO + ly, gx = tx - HALO + lx;
            float v = (gy >= 0 && gy < H && gx >= 0 && gx < W) ? fp[gy * W + gx] : 0.f;
            fA[i] = v * sm[i];
        }
        __syncthreads();
        float* src = fA; float* dst = fB;
        for (int s = 0; s < 5; ++s) {
            for (int i = tid; i < TW * TW; i += nth) {
                int ly = i / TW, lx = i % TW;
                float v = 0.f;
                if (ly > 0 && ly < TW - 1 && lx > 0 && lx < TW - 1) {
                    float nb = 0.25f * (src[i - TW] + src[i + TW] + src[i - 1] + src[i + 1]);
                    v = sm[i] * (0.5f * src[i] + 0.5f * nb / sc[i]);
                }
                dst[i] = v;
            }
            __syncthreads();
            float* t = src; src = dst; dst = t;
        }
        float* meanp = mean + ((long)(b * C + c)) * HW;
        float* m2p   = m2   + ((long)(b * C + c)) * HW;
        for (int i = tid; i < TS * TS; i += nth) {
            int ly = HALO + i / TS, lx = HALO + i % TS;
            int gy = ty + i / TS, gx = tx + i % TS;
            if (gy < H && gx < W) {
                float comp = src[ly * TW + lx];
                long g = (long)gy * W + gx;
                float mo = meanp[g];
                float d = comp - mo;
                float mn = mo + d / count;
                m2p[g] += d * (comp - mn);
                meanp[g] = mn;
            }
        }
        __syncthreads();
    }
}

__global__ void stability_kernel(const float* __restrict__ m2, const float* __restrict__ birth,
                                 float* __restrict__ stab, float* __restrict__ persist,
                                 int C, int HW, long BHW, float denom)
{
    for (long i = blockIdx.x * (long)blockDim.x + threadIdx.x; i < BHW;
         i += (long)gridDim.x * blockDim.x) {
        long b = i / HW, p = i % HW;
        const float* mp = m2 + (long)b * C * HW + p;
        float s = 0.f;
        for (int c = 0; c < C; ++c)
            s += mp[(long)c * HW];
        float st = 1.f / (1.f + s / denom);
        stab[i] = st;
        persist[i] = (1.f - birth[i]) * st;
    }
}

__global__ void scorein_kernel(const float* __restrict__ feats, const float* __restrict__ persist,
                               const float* __restrict__ birth, const float* __restrict__ stab,
                               float* __restrict__ out, int C, int HW, long total)
{
    int Ct = C + 3;
    for (long i = blockIdx.x * (long)blockDim.x + threadIdx.x; i < total;
         i += (long)gridDim.x * blockDim.x) {
        long b = i / ((long)Ct * HW);
        int c = (int)((i / HW) % Ct);
        long p = i % HW;
        float v;
        if (c < C)           v = feats[((long)(b * C + c)) * HW + p];
        else if (c == C)     v = persist[b * HW + p];
        else if (c == C + 1) v = birth[b * HW + p];
        else                 v = stab[b * HW + p];
        out[i] = v;
    }
}

// top-K by iterative argmax (K=100, N=25600 per batch); one block per batch.
__global__ void topk_kernel(const float* __restrict__ scores, float* __restrict__ scratch,
                            float* __restrict__ topv, int* __restrict__ topi, int HW, int Kp)
{
    int b = blockIdx.x, tid = threadIdx.x;
    const float* s = scores + (long)b * HW;
    float* scp = scratch + (long)b * HW;
    for (int i = tid; i < HW; i += blockDim.x) scp[i] = s[i];
    __syncthreads();
    __shared__ float sv[256];
    __shared__ int si[256];
    for (int k = 0; k < Kp; ++k) {
        float bv = -3.4e38f; int bi = 0;
        for (int i = tid; i < HW; i += blockDim.x) {
            float v = scp[i];
            if (v > bv || (v == bv && i < bi)) { bv = v; bi = i; }
        }
        sv[tid] = bv; si[tid] = bi; __syncthreads();
        for (int off = 128; off > 0; off >>= 1) {
            if (tid < off) {
                if (sv[tid + off] > sv[tid] ||
                    (sv[tid + off] == sv[tid] && si[tid + off] < si[tid])) {
                    sv[tid] = sv[tid + off]; si[tid] = si[tid + off];
                }
            }
            __syncthreads();
        }
        if (tid == 0) {
            topv[b * Kp + k] = sv[0];
            topi[b * Kp + k] = si[0];
            scp[si[0]] = -3.4e38f;
        }
        __syncthreads();
    }
}

__global__ void gather_kernel(const float* __restrict__ feats, const int* __restrict__ topi,
                              float* __restrict__ localf, int C, int HW, int Kp, long total)
{
    for (long i = blockIdx.x * (long)blockDim.x + threadIdx.x; i < total;
         i += (long)gridDim.x * blockDim.x) {
        long n = i / C; int c = (int)(i % C);
        long b = n / Kp;
        int idx = topi[n];
        localf[i] = feats[((long)(b * C + c)) * HW + idx];
    }
}

__global__ void confin_kernel(const float* __restrict__ feat, const float* __restrict__ topv,
                              float* __restrict__ out, long total)
{
    for (long i = blockIdx.x * (long)blockDim.x + threadIdx.x; i < total;
         i += (long)gridDim.x * blockDim.x) {
        long n = i / 129; int c = (int)(i % 129);
        out[i] = (c < 128) ? feat[n * 128 + c] : topv[n];
    }
}

__global__ void finish_heads_kernel(const int* __restrict__ topi,
                                    const float* __restrict__ t2a, const float* __restrict__ t2b,
                                    const float* __restrict__ t1a, const float* __restrict__ t1b,
                                    float* __restrict__ centers, float* __restrict__ wh,
                                    float* __restrict__ ang, float* __restrict__ conf,
                                    int W, int H, int total)
{
    int n = blockIdx.x * blockDim.x + threadIdx.x;
    if (n >= total) return;
    int idx = topi[n];
    float xs = (float)(idx % W) / (float)W;
    float ys = (float)(idx / W) / (float)H;
    float ox = tanhf(t2a[n * 2 + 0]);
    float oy = tanhf(t2a[n * 2 + 1]);
    centers[n * 2 + 0] = (xs + ox / (float)W) * 640.f;
    centers[n * 2 + 1] = (ys + oy / (float)H) * 640.f;
    wh[n * 2 + 0] = expf(fminf(fmaxf(t2b[n * 2 + 0], -5.f), 5.f)) * 64.f;
    wh[n * 2 + 1] = expf(fminf(fmaxf(t2b[n * 2 + 1], -5.f), 5.f)) * 64.f;
    float a = t1a[n];
    ang[n] = atan2f(sinf(a), cosf(a));
    conf[n] = 1.f / (1.f + expf(-t1b[n]));
}

__global__ void copymaps_kernel(const float* __restrict__ persist, const float* __restrict__ filt,
                                float* __restrict__ o1, float* __restrict__ o2, long n)
{
    for (long i = blockIdx.x * (long)blockDim.x + threadIdx.x; i < n;
         i += (long)gridDim.x * blockDim.x) {
        o1[i] = persist[i];
        o2[i] = filt[i];
    }
}

// ---------------------------------------------------------------------------
// Host-side launch helpers
// ---------------------------------------------------------------------------
static inline int gblocks(long n, int t)
{
    long b = (n + t - 1) / t;
    if (b > 262144) b = 262144;
    if (b < 1) b = 1;
    return (int)b;
}

static void launch_conv(const float* in, const float* w, float* wpack,
                        const float* bias, float* out,
                        int B, int Cin, int H, int W, int Cout, int OH, int OW,
                        int stride, int act, hipStream_t s)
{
    int Mtiles = (Cout + 15) / 16;
    int Ntiles = (B * OH * OW + 15) / 16;
    int CoutP = Mtiles * 16;
    int CinP = (Cin + 31) & ~31;
    _Float16* w2 = (_Float16*)wpack;
    long rtotal = (long)CoutP * 9 * CinP;
    repack_w_kernel<<<gblocks(rtotal, 256), 256, 0, s>>>(w, w2, Cout, Cin, CinP, rtotal);
    long waves = (long)Mtiles * Ntiles;
    int blocks = (int)((waves + 7) / 8);
    conv3x3_wmma_kernel<<<blocks, 256, 0, s>>>(in, w2, bias, out, B, Cin, CinP, H, W,
                                               Cout, OH, OW, stride, act,
                                               Mtiles, Ntiles);
}

static void launch_gemm(const float* X, const float* Wm, const float* bias, float* out,
                        int N, int K, int M, int act, hipStream_t s)
{
    int Mtiles = (M + 15) / 16;
    int Ntiles = (N + 15) / 16;
    long waves = (long)Mtiles * Ntiles;
    int blocks = (int)((waves + 7) / 8);
    gemm_wmma_kernel<<<blocks, 256, 0, s>>>(X, Wm, bias, out, N, K, M, act, Mtiles, Ntiles);
}

extern "C" void kernel_launch(void* const* d_in, const int* in_sizes, int n_in,
                              void* d_out, int out_size, void* d_ws, size_t ws_size,
                              hipStream_t stream)
{
    (void)in_sizes; (void)n_in; (void)out_size; (void)ws_size;

    const float* images = (const float*)d_in[0];
    const float* e1w = (const float*)d_in[1];  const float* e1b  = (const float*)d_in[2];
    const float* e1g = (const float*)d_in[3];  const float* e1be = (const float*)d_in[4];
    const float* e2w = (const float*)d_in[5];  const float* e2b  = (const float*)d_in[6];
    const float* e2g = (const float*)d_in[7];  const float* e2be = (const float*)d_in[8];
    const float* e3w = (const float*)d_in[9];  const float* e3b  = (const float*)d_in[10];
    const float* e3g = (const float*)d_in[11]; const float* e3be = (const float*)d_in[12];
    const float* fw1 = (const float*)d_in[13]; const float* fb1  = (const float*)d_in[14];
    const float* fw2 = (const float*)d_in[15]; const float* fb2  = (const float*)d_in[16];
    const float* hww = (const float*)d_in[17]; const float* hb   = (const float*)d_in[18];
    const float* hg  = (const float*)d_in[19]; const float* hbe  = (const float*)d_in[20];
    const float* lvo = (const float*)d_in[21];
    const float* sw1 = (const float*)d_in[22]; const float* sb1  = (const float*)d_in[23];
    const float* sw2 = (const float*)d_in[24]; const float* sb2  = (const float*)d_in[25];
    const float* dw  = (const float*)d_in[26]; const float* db   = (const float*)d_in[27];
    const float* cw1 = (const float*)d_in[28]; const float* cb1  = (const float*)d_in[29];
    const float* cw2 = (const float*)d_in[30]; const float* cb2  = (const float*)d_in[31];
    const float* zw1 = (const float*)d_in[32]; const float* zb1  = (const float*)d_in[33];
    const float* zw2 = (const float*)d_in[34]; const float* zb2  = (const float*)d_in[35];
    const float* aw1 = (const float*)d_in[36]; const float* ab1  = (const float*)d_in[37];
    const float* aw2 = (const float*)d_in[38]; const float* ab2  = (const float*)d_in[39];
    const float* kw1 = (const float*)d_in[40]; const float* kb1  = (const float*)d_in[41];
    const float* kw2 = (const float*)d_in[42]; const float* kb2  = (const float*)d_in[43];
    const float* qw1 = (const float*)d_in[44]; const float* qb1  = (const float*)d_in[45];
    const float* qw2 = (const float*)d_in[46]; const float* qb2  = (const float*)d_in[47];

    const int B = 4, C = 128, H = 160, W = 160;
    const long HW = (long)H * W;          // 25600
    const long BHW = B * HW;              // 102400
    const long SZ1 = 4L * 32 * 320 * 320; // x1 / mean
    const long SZ2 = 4L * 64 * HW;        // x2 / f1 / sconv
    const long SZ3 = 4L * 131 * HW;       // enc / score_in
    const long SZ4 = 4L * 128 * HW;       // features / m2
    const int  KP = 100, NR = B * KP;     // 400 proposal rows

    float* ws = (float*)d_ws;
    float* buf1 = ws;                     // x1, later Welford mean
    float* buf2 = buf1 + SZ1;             // x2, later fw1/sw1 output
    float* buf3 = buf2 + SZ2;             // enc, later score_in (131 ch)
    float* buf4 = buf3 + SZ3;             // features
    float* m2b  = buf4 + SZ4;             // Welford m2
    float* sm   = m2b + SZ4;              // small region (16B-aligned here)
    float* wpack   = sm;            sm += 131072;  // repacked f16 weights (<=288KB used of 512KB)
    float* filt    = sm;            sm += BHW;
    float* maskb   = sm;            sm += BHW;   // also top-k scratch
    float* birth   = sm;            sm += BHW;
    float* stab    = sm;            sm += BHW;
    float* persist = sm;            sm += BHW;
    float* scores  = sm;            sm += BHW;
    float* stats   = sm;            sm += 256;
    float* lev     = sm;            sm += 16;
    float* fmn     = sm;            sm += 4;
    float* fmx     = sm;            sm += 4;
    float* localf  = sm;            sm += (long)NR * 128;
    float* dfeat   = sm;            sm += (long)NR * 128;
    float* h64     = sm;            sm += (long)NR * 64;
    float* h32     = sm;            sm += (long)NR * 32;
    float* h16     = sm;            sm += (long)NR * 16;
    float* t2a     = sm;            sm += (long)NR * 2;
    float* t2b     = sm;            sm += (long)NR * 2;
    float* t1a     = sm;            sm += NR;
    float* t1b     = sm;            sm += NR;
    float* confin  = sm;            sm += (long)NR * 129;
    int*   topi    = (int*)sm;      sm += NR;

    float* out = (float*)d_out;
    float* o_centers = out;              // 800
    float* o_wh      = out + 800;        // 800
    float* o_ang     = out + 1600;       // 400
    float* o_cls     = out + 2000;       // 6000
    float* o_conf    = out + 8000;       // 400
    float* o_topv    = out + 8400;       // 400
    float* o_pers    = out + 8800;       // 102400
    float* o_filt    = out + 111200;     // 102400

    // ---- encoder ----
    launch_conv(images, e1w, wpack, e1b, buf1, B, 3, 640, 640, 32, 320, 320, 2, 0, stream);
    bn_stats_kernel<<<32, 256, 0, stream>>>(buf1, stats, B, 32, 320 * 320);
    bn_apply_kernel<<<gblocks(SZ1, 256), 256, 0, stream>>>(buf1, stats, e1g, e1be,
                                                           32, 320 * 320, SZ1, 1);
    launch_conv(buf1, e2w, wpack, e2b, buf2, B, 32, 320, 320, 64, 160, 160, 2, 0, stream);
    bn_stats_kernel<<<64, 256, 0, stream>>>(buf2, stats, B, 64, (int)HW);
    bn_apply_kernel<<<gblocks(SZ2, 256), 256, 0, stream>>>(buf2, stats, e2g, e2be,
                                                           64, (int)HW, SZ2, 1);
    launch_conv(buf2, e3w, wpack, e3b, buf3, B, 64, 160, 160, 128, 160, 160, 1, 0, stream);
    bn_stats_kernel<<<128, 256, 0, stream>>>(buf3, stats, B, 128, (int)HW);
    bn_apply_kernel<<<gblocks(SZ4, 256), 256, 0, stream>>>(buf3, stats, e3g, e3be,
                                                           128, (int)HW, SZ4, 1);

    // ---- filtration head: relu(conv3x3 128->64), then 1x1 64->1 ----
    launch_conv(buf3, fw1, wpack, fb1, buf2, B, 128, 160, 160, 64, 160, 160, 1, 1, stream);
    conv1x1_kernel<<<gblocks(BHW, 256), 256, 0, stream>>>(buf2, fw2, fb2, filt,
                                                          64, (int)HW, BHW, 0);

    // ---- features: gelu(bn(conv3x3 128->128)) ----
    launch_conv(buf3, hww, wpack, hb, buf4, B, 128, 160, 160, 128, 160, 160, 1, 0, stream);
    bn_stats_kernel<<<128, 256, 0, stream>>>(buf4, stats, B, 128, (int)HW);
    bn_apply_kernel<<<gblocks(SZ4, 256), 256, 0, stream>>>(buf4, stats, hg, hbe,
                                                           128, (int)HW, SZ4, 1);

    // ---- filtration range + levels ----
    minmax_kernel<<<B, 256, 0, stream>>>(filt, fmn, fmx, (int)HW);
    levels_kernel<<<1, 32, 0, stream>>>(lvo, lev, 16);

    // ---- reset scan state (determinism across graph replays) ----
    fill_kernel<<<gblocks(BHW, 256), 256, 0, stream>>>(maskb, 0.f, BHW);
    fill_kernel<<<gblocks(BHW, 256), 256, 0, stream>>>(birth, 0.f, BHW);
    fill_kernel<<<gblocks(SZ4, 256), 256, 0, stream>>>(buf1, 0.f, SZ4);  // mean
    fill_kernel<<<gblocks(SZ4, 256), 256, 0, stream>>>(m2b, 0.f, SZ4);   // m2

    // ---- 16-level filtration scan with fused 5-step diffusion ----
    const int tiles = ((W + 31) / 32) * ((H + 31) / 32);   // 25
    for (int l = 0; l < 16; ++l) {
        mask_birth_kernel<<<gblocks(BHW, 256), 256, 0, stream>>>(
            filt, maskb, birth, lev, fmn, fmx, l, (int)HW, BHW);
        diffusion_kernel<<<dim3(tiles, B, 8), 256, 0, stream>>>(
            maskb, buf4, buf1, m2b, C, H, W, (float)(l + 1), 16);
    }

    // ---- stability / persistence ----
    stability_kernel<<<gblocks(BHW, 256), 256, 0, stream>>>(
        m2b, birth, stab, persist, C, (int)HW, BHW, 15.f * 128.f);

    // ---- score head on concat(features, persist, birth, stab) = 131 ch ----
    scorein_kernel<<<gblocks(SZ3, 256), 256, 0, stream>>>(
        buf4, persist, birth, stab, buf3, 128, (int)HW, SZ3);
    launch_conv(buf3, sw1, wpack, sb1, buf2, B, 131, 160, 160, 64, 160, 160, 1, 1, stream);
    conv1x1_kernel<<<gblocks(BHW, 256), 256, 0, stream>>>(buf2, sw2, sb2, scores,
                                                          64, (int)HW, BHW, 1);

    // ---- top-K proposals ----
    topk_kernel<<<B, 256, 0, stream>>>(scores, maskb, o_topv, topi, (int)HW, KP);
    gather_kernel<<<gblocks((long)NR * 128, 256), 256, 0, stream>>>(
        buf4, topi, localf, 128, (int)HW, KP, (long)NR * 128);

    // ---- proposal MLP heads (WMMA GEMMs over 400 rows) ----
    launch_gemm(localf, dw, db, dfeat, NR, 128, 128, 1, stream);        // feat
    launch_gemm(dfeat, cw1, cb1, h32, NR, 128, 32, 1, stream);
    launch_gemm(h32, cw2, cb2, t2a, NR, 32, 2, 0, stream);              // center offsets
    launch_gemm(dfeat, zw1, zb1, h32, NR, 128, 32, 1, stream);
    launch_gemm(h32, zw2, zb2, t2b, NR, 32, 2, 0, stream);              // wh raw
    launch_gemm(dfeat, aw1, ab1, h16, NR, 128, 16, 1, stream);
    launch_gemm(h16, aw2, ab2, t1a, NR, 16, 1, 0, stream);              // angle raw
    launch_gemm(dfeat, kw1, kb1, h64, NR, 128, 64, 1, stream);
    launch_gemm(h64, kw2, kb2, o_cls, NR, 64, 15, 0, stream);           // cls logits
    confin_kernel<<<gblocks((long)NR * 129, 256), 256, 0, stream>>>(
        dfeat, o_topv, confin, (long)NR * 129);
    launch_gemm(confin, qw1, qb1, h16, NR, 129, 16, 1, stream);
    launch_gemm(h16, qw2, qb2, t1b, NR, 16, 1, 0, stream);              // conf raw

    // ---- finishers ----
    finish_heads_kernel<<<(NR + 255) / 256, 256, 0, stream>>>(
        topi, t2a, t2b, t1a, t1b, o_centers, o_wh, o_ang, o_conf, W, H, NR);
    copymaps_kernel<<<gblocks(BHW, 256), 256, 0, stream>>>(persist, filt, o_pers, o_filt, BHW);
}
